// Encoder_44040594653903
// MI455X (gfx1250) — compile-verified, matrix operand order checked by reading, not run on hardware
//
#include <hip/hip_runtime.h>

#define S_LEN  4096
#define HDIM   256
#define NHEADS 6
#define GDIM   (3*HDIM)        // 768
#define OCAT   (NHEADS*HDIM)   // 1536
#define LN_EPS 1e-5f

typedef __bf16 bf16;
typedef __attribute__((ext_vector_type(16))) __bf16 bf16x16;
typedef __attribute__((ext_vector_type(8)))  float  f32x8;

// ---------------------------------------------------------------- WMMA helpers
__device__ __forceinline__ f32x8 wmma_bf(bf16x16 a, bf16x16 b, f32x8 c) {
  return __builtin_amdgcn_wmma_f32_16x16x32_bf16(false, a, false, b, (short)0, c,
                                                 false, false);
}

// Fragment-major ("packed") operand layout: a 16x32 A-tile / 32x16 B-tile is
// stored as 32 lanes x 16 bf16 (1 KB); a per-lane fragment load is 32
// contiguous 32B-aligned bytes -> 2x b128.  Within-tile position of element
// (rr = m&15 for A / n&15 for B, kk = k&31):
//   lane = rr | (((kk>>3)&1) << 4)
//   elem = (((kk>>4)<<2) + ((kk&7)>>1))*2 + (kk&1)     (ISA 7.12.2)
__device__ __forceinline__ int frag_elem(int kk) {
  return (((kk >> 4) << 2) + ((kk & 7) >> 1)) * 2 + (kk & 1);
}
__device__ __forceinline__ int frag_lane(int rr, int kk) {
  return rr | (((kk >> 3) & 1) << 4);
}
__device__ __forceinline__ bf16x16 load_frag(const bf16* p) {
  return *(const bf16x16*)p;     // 2x (global|ds)_load_b128
}

// CDNA5 async memory->LDS (no VGPR round-trip, tracked by ASYNCcnt).
__device__ __forceinline__ void async_b128(void* lds_dst, const void* src) {
  unsigned off = (unsigned)(size_t)lds_dst;   // low 32 bits = LDS offset
  asm volatile("global_load_async_to_lds_b128 %0, %1, off"
               :: "v"(off), "v"(src) : "memory");
}
__device__ __forceinline__ void wait_async() {
  asm volatile("s_wait_asynccnt 0x0" ::: "memory");
}

// ---------------------------------------------------------------- small utils
__global__ void embed_kernel(const int* __restrict__ tokens,
                             const float* __restrict__ emb,
                             float* __restrict__ x, bf16* __restrict__ xb) {
  int s = blockIdx.x, d = threadIdx.x;
  float v = emb[(long)tokens[s]*HDIM + d];
  x[(long)s*HDIM + d] = v;
  size_t t = (size_t)(s >> 4) * (HDIM >> 5) + (d >> 5);       // packed-A
  xb[(t*32 + frag_lane(s & 15, d & 31))*16 + frag_elem(d & 31)] = (bf16)v;
}

// Pack a KxN f32 matrix (optionally transposed view) into B-fragment layout,
// tile id = (k>>5)*(N/16) + (n>>4). grid.z = batch.
__global__ void pack_b_kernel(const float* __restrict__ src, bf16* __restrict__ dst,
                              int K, int N, int transpose,
                              long srcBatch, long dstBatch) {
  const float* s = src + (long)blockIdx.z * srcBatch;
  bf16* d = dst + (long)blockIdx.z * dstBatch;
  long idx = (long)blockIdx.x * 256 + threadIdx.x;
  int k = (int)(idx / N), n = (int)(idx % N);
  float v = transpose ? s[(long)n*K + k] : s[(long)k*N + n];
  size_t t = (size_t)(k >> 5) * (N >> 4) + (n >> 4);
  d[(t*32 + frag_lane(n & 15, k & 31))*16 + frag_elem(k & 31)] = (bf16)v;
}

// ---------------------------------------------------------------- batched GEMM
// C[b] = A[b](MxK packed-A) * B[b](KxN packed-B) + bias[b].  64x64 tile /
// 128-thread WG.  Software-pipelined: fragments for kt+1 are in flight while
// the WMMAs for kt execute (partial s_wait_loadcnt instead of full drains).
// pack_mode: 0=f32 row-major C; 1=bf16 rowpack (Q/K); 2=bf16 colpack (V).
__global__ void gemm_packed_kernel(const bf16* __restrict__ A,
                                   const bf16* __restrict__ B,
                                   const float* __restrict__ bias,
                                   void* __restrict__ Cout,
                                   int M, int N, int K,
                                   long batchA, long batchB, long batchBias,
                                   long batchC, int pack_mode) {
  const int b = blockIdx.z;
  const bf16* Ab = A + (long)b * batchA;
  const bf16* Bb = B + (long)b * batchB;
  const float* bi = bias ? bias + (long)b * batchBias : nullptr;
  const int wave = threadIdx.x >> 5, lane = threadIdx.x & 31;
  const int row0 = blockIdx.y * 64 + wave * 16;
  const int col0 = blockIdx.x * 64;
  const int KT = K >> 5, NT = N >> 4, nt0 = col0 >> 4;
  const size_t abase = (size_t)(row0 >> 4) * KT;

  f32x8 acc[4] = {};
  bf16x16 ac = load_frag(Ab + ((abase + 0)*32 + lane)*16);
  bf16x16 bc[4];
#pragma unroll
  for (int j = 0; j < 4; ++j)
    bc[j] = load_frag(Bb + ((size_t)(0*NT + nt0 + j)*32 + lane)*16);

  for (int kt = 0; kt < KT - 1; ++kt) {
    bf16x16 an = load_frag(Ab + ((abase + kt + 1)*32 + lane)*16);
    bf16x16 bn[4];
#pragma unroll
    for (int j = 0; j < 4; ++j)
      bn[j] = load_frag(Bb + ((size_t)((kt + 1)*NT + nt0 + j)*32 + lane)*16);
#pragma unroll
    for (int j = 0; j < 4; ++j) acc[j] = wmma_bf(ac, bc[j], acc[j]);
    ac = an;
#pragma unroll
    for (int j = 0; j < 4; ++j) bc[j] = bn[j];
  }
#pragma unroll
  for (int j = 0; j < 4; ++j) acc[j] = wmma_bf(ac, bc[j], acc[j]);

#pragma unroll
  for (int j = 0; j < 4; ++j)
#pragma unroll
    for (int v = 0; v < 8; ++v) {
      int r = row0 + ((lane >> 4) << 3) + v;
      int c = col0 + j*16 + (lane & 15);
      float val = acc[j][v] + (bi ? bi[c] : 0.0f);
      if (pack_mode == 0) {
        ((float*)Cout)[(long)b*batchC + (long)r*N + c] = val;
      } else if (pack_mode == 1) {     // rowpack: tile=(r>>4)*(N/32)+(c>>5)
        size_t t = (size_t)(r >> 4) * (N >> 5) + (c >> 5);
        ((bf16*)Cout)[(long)b*batchC +
                      (long)((t*32 + frag_lane(r & 15, c & 31))*16 +
                             frag_elem(c & 31))] = (bf16)val;
      } else {                          // colpack: tile=(r>>5)*(N/16)+(c>>4)
        size_t t = (size_t)(r >> 5) * (N >> 4) + (c >> 4);
        ((bf16*)Cout)[(long)b*batchC +
                      (long)((t*32 + frag_lane(c & 15, r & 31))*16 +
                             frag_elem(r & 31))] = (bf16)val;
      }
    }
}

// ---------------------------------------------------------------- flash attention
// grid (S/64, NH), 128 threads. Q/K/V pre-packed so each 64-row block is a
// contiguous 32KB region; staging uses async memory->LDS; fragment reads are
// b128 pairs; score loop software-pipelined; online softmax (no SxS matrix).
__global__ void attn_kernel(const bf16* __restrict__ Qp,
                            const bf16* __restrict__ Kp,
                            const bf16* __restrict__ Vp,
                            bf16* __restrict__ Oc) {
  extern __shared__ char smem[];
  bf16* Qs = (bf16*)smem;            // 32 tiles * 512 = 16384 elems (32 KB)
  bf16* Ks = Qs + 16384;
  bf16* Vs = Ks + 16384;
  bf16* Ps = Vs + 16384;             // 4 waves * 2 tiles * 512 elems

  const int head = blockIdx.y;
  const int q0   = blockIdx.x * 64;
  const bf16* Qg = Qp + (size_t)head*S_LEN*HDIM + (size_t)q0*HDIM;
  const bf16* Kg = Kp + (size_t)head*S_LEN*HDIM;
  const bf16* Vg = Vp + (size_t)head*S_LEN*HDIM;
  const int tid = threadIdx.x, wave = tid >> 5, lane = tid & 31;

  for (int i = tid; i < 2048; i += 128)        // async Q block -> LDS
    async_b128((uint4*)Qs + i, (const uint4*)Qg + i);

  f32x8 o[16] = {};
  float m_run[8], l_run[8];
#pragma unroll
  for (int v = 0; v < 8; ++v) { m_run[v] = -1e30f; l_run[v] = 0.0f; }

  const float scale = 0.0625f;       // 1/sqrt(256)
  bf16* Pw = Ps + wave*1024;

  for (int kv0 = 0; kv0 < S_LEN; kv0 += 64) {
    __syncthreads();                 // previous iter done with Ks/Vs
    for (int i = tid; i < 2048; i += 128) {
      async_b128((uint4*)Ks + i, (const uint4*)(Kg + (size_t)kv0*HDIM) + i);
      async_b128((uint4*)Vs + i, (const uint4*)(Vg + (size_t)kv0*HDIM) + i);
    }
    if (kv0 + 64 < S_LEN) {          // pull next block toward L2/WGP$
      __builtin_prefetch(Kg + (size_t)(kv0 + 64)*HDIM + (size_t)tid*128, 0, 1);
      __builtin_prefetch(Vg + (size_t)(kv0 + 64)*HDIM + (size_t)tid*128, 0, 1);
    }
    wait_async();                    // also covers the Q copy on iter 0
    __syncthreads();

    // scores: Q_w (16x256) * K_tile^T -> 16x64, pipelined over k-tiles
    f32x8 sc[4] = {};
    bf16x16 ac = load_frag(Qs + ((size_t)(wave*8 + 0)*32 + lane)*16);
    bf16x16 bc[4];
#pragma unroll
    for (int j = 0; j < 4; ++j)
      bc[j] = load_frag(Ks + ((size_t)(j*8 + 0)*32 + lane)*16);
#pragma unroll
    for (int kt = 0; kt < 7; ++kt) {
      bf16x16 an = load_frag(Qs + ((size_t)(wave*8 + kt + 1)*32 + lane)*16);
      bf16x16 bn[4];
#pragma unroll
      for (int j = 0; j < 4; ++j)
        bn[j] = load_frag(Ks + ((size_t)(j*8 + kt + 1)*32 + lane)*16);
#pragma unroll
      for (int j = 0; j < 4; ++j) sc[j] = wmma_bf(ac, bc[j], sc[j]);
      ac = an;
#pragma unroll
      for (int j = 0; j < 4; ++j) bc[j] = bn[j];
    }
#pragma unroll
    for (int j = 0; j < 4; ++j) sc[j] = wmma_bf(ac, bc[j], sc[j]);

    // online softmax: row r = (lane>>4)*8+v lives in 16 lanes of one half
#pragma unroll
    for (int v = 0; v < 8; ++v) {
      float mx = -1e30f;
#pragma unroll
      for (int j = 0; j < 4; ++j) { sc[j][v] *= scale; mx = fmaxf(mx, sc[j][v]); }
      for (int off = 1; off < 16; off <<= 1) mx = fmaxf(mx, __shfl_xor(mx, off, 32));
      float mnew  = fmaxf(m_run[v], mx);
      float alpha = __expf(m_run[v] - mnew);
      float rsum  = 0.0f;
#pragma unroll
      for (int j = 0; j < 4; ++j) {
        float pv = __expf(sc[j][v] - mnew);
        sc[j][v] = pv; rsum += pv;
      }
      for (int off = 1; off < 16; off <<= 1) rsum += __shfl_xor(rsum, off, 32);
      l_run[v] = l_run[v]*alpha + rsum;
      m_run[v] = mnew;
#pragma unroll
      for (int j2 = 0; j2 < 16; ++j2) o[j2][v] *= alpha;
    }

    // scatter P into packed-A layout in per-wave LDS (16x64, 2 k-tiles)
#pragma unroll
    for (int j = 0; j < 4; ++j)
#pragma unroll
      for (int v = 0; v < 8; ++v) {
        int r = ((lane >> 4) << 3) + v, c = j*16 + (lane & 15);
        Pw[((size_t)(c >> 5)*32 + frag_lane(r & 15, c & 31))*16 +
           frag_elem(c & 31)] = (bf16)sc[j][v];
      }

    // O += P (16x64) * V_tile (64x256); pipelined over d-tiles
#pragma unroll
    for (int kt = 0; kt < 2; ++kt) {
      bf16x16 af = load_frag(Pw + ((size_t)kt*32 + lane)*16);
      bf16x16 bcur = load_frag(Vs + ((size_t)(kt*16 + 0)*32 + lane)*16);
#pragma unroll
      for (int j = 0; j < 16; ++j) {
        int jn = (j + 1) & 15;
        bf16x16 bnxt = load_frag(Vs + ((size_t)(kt*16 + jn)*32 + lane)*16);
        o[j] = wmma_bf(af, bcur, o[j]);
        bcur = bnxt;
      }
    }
  }

  // epilogue: normalize, write into packed-A layout of [S, NH*H] (K = 1536)
#pragma unroll
  for (int j = 0; j < 16; ++j)
#pragma unroll
    for (int v = 0; v < 8; ++v) {
      int r  = q0 + wave*16 + ((lane >> 4) << 3) + v;
      int cc = head*HDIM + j*16 + (lane & 15);
      float val = o[j][v] / l_run[v];
      size_t t = (size_t)(r >> 4) * (OCAT >> 5) + (cc >> 5);
      Oc[(t*32 + frag_lane(r & 15, cc & 31))*16 + frag_elem(cc & 31)] = (bf16)val;
    }
}

// ---------------------------------------------------------------- residual + LN
__global__ void ln_kernel(const float* __restrict__ x, const float* __restrict__ op,
                          const float* __restrict__ g, const float* __restrict__ b,
                          bf16* __restrict__ yb) {
  __shared__ float red[8];
  const int s = blockIdx.x, d = threadIdx.x;
  float v = x[(long)s*HDIM + d] + op[(long)s*HDIM + d];

  float t = v;
  for (int off = 16; off; off >>= 1) t += __shfl_xor(t, off, 32);
  if ((d & 31) == 0) red[d >> 5] = t;
  __syncthreads();
  float mu = 0.0f;
#pragma unroll
  for (int i = 0; i < 8; ++i) mu += red[i];
  mu *= (1.0f / HDIM);
  __syncthreads();

  float dv = v - mu;
  t = dv * dv;
  for (int off = 16; off; off >>= 1) t += __shfl_xor(t, off, 32);
  if ((d & 31) == 0) red[d >> 5] = t;
  __syncthreads();
  float var = 0.0f;
#pragma unroll
  for (int i = 0; i < 8; ++i) var += red[i];
  var *= (1.0f / HDIM);

  float y = dv * rsqrtf(var + LN_EPS) * g[d] + b[d];
  size_t tt = (size_t)(s >> 4) * (HDIM >> 5) + (d >> 5);   // packed-A
  yb[(tt*32 + frag_lane(s & 15, d & 31))*16 + frag_elem(d & 31)] = (bf16)y;
}

// ---------------------------------------------------------------- sequential GRU
__global__ void gru_kernel(const float* __restrict__ GI,
                           const float* __restrict__ Whh,
                           const float* __restrict__ bhh,
                           float* __restrict__ out) {
  __shared__ float h[HDIM];
  __shared__ float gh[GDIM];
  const int i = threadIdx.x;                 // 0..767
  if (i < HDIM) h[i] = 0.0f;
  const float4* Wrow = (const float4*)(Whh + (long)i*HDIM);
  const float bh = bhh[i];
  __syncthreads();

  for (int t = 0; t < S_LEN; ++t) {
    float acc = bh;
#pragma unroll 8
    for (int k = 0; k < HDIM/4; ++k) {
      float4 w = Wrow[k];
      acc += w.x*h[4*k] + w.y*h[4*k+1] + w.z*h[4*k+2] + w.w*h[4*k+3];
    }
    gh[i] = acc;
    __syncthreads();
    if (i < HDIM) {
      const float* git = GI + (long)t*GDIM;
      float r = 1.0f / (1.0f + __expf(-(git[i]      + gh[i])));
      float z = 1.0f / (1.0f + __expf(-(git[HDIM+i] + gh[HDIM+i])));
      float n = tanhf(git[2*HDIM+i] + r*gh[2*HDIM+i]);
      h[i] = (1.0f - z)*n + z*h[i];
    }
    __syncthreads();
  }
  if (i < HDIM) { out[i] = h[i]; out[HDIM + i] = h[i]; }
}

// ---------------------------------------------------------------- launcher
extern "C" void kernel_launch(void* const* d_in, const int* in_sizes, int n_in,
                              void* d_out, int out_size, void* d_ws, size_t ws_size,
                              hipStream_t stream) {
  (void)in_sizes; (void)n_in; (void)out_size; (void)ws_size;
  const int*   tokens = (const int*)  d_in[0];
  const float* emb    = (const float*)d_in[1];
  const float* Wq     = (const float*)d_in[2];
  const float* bq     = (const float*)d_in[3];
  const float* Wk     = (const float*)d_in[4];
  const float* bk     = (const float*)d_in[5];
  const float* Wv     = (const float*)d_in[6];
  const float* bv     = (const float*)d_in[7];
  const float* Wo     = (const float*)d_in[8];
  const float* bo     = (const float*)d_in[9];
  const float* ln_g   = (const float*)d_in[10];
  const float* ln_b   = (const float*)d_in[11];
  const float* W_ih   = (const float*)d_in[12];
  const float* W_hh   = (const float*)d_in[13];
  const float* b_ih   = (const float*)d_in[14];
  const float* b_hh   = (const float*)d_in[15];

  char* p = (char*)d_ws;
  auto carve = [&](size_t bytes) {
    char* r = p; p += (bytes + 255) & ~(size_t)255; return r;
  };
  float* x     = (float*)carve((size_t)S_LEN*HDIM*4);
  bf16*  xb    = (bf16*) carve((size_t)S_LEN*HDIM*2);           // packed-A
  bf16*  Wq_b  = (bf16*) carve((size_t)NHEADS*HDIM*HDIM*2);     // packed-B
  bf16*  Wk_b  = (bf16*) carve((size_t)NHEADS*HDIM*HDIM*2);
  bf16*  Wv_b  = (bf16*) carve((size_t)NHEADS*HDIM*HDIM*2);
  bf16*  Wo_b  = (bf16*) carve((size_t)OCAT*HDIM*2);
  bf16*  WihT  = (bf16*) carve((size_t)HDIM*GDIM*2);
  bf16*  Qb    = (bf16*) carve((size_t)NHEADS*S_LEN*HDIM*2);    // rowpack
  bf16*  Kb    = (bf16*) carve((size_t)NHEADS*S_LEN*HDIM*2);    // rowpack
  bf16*  Vb    = (bf16*) carve((size_t)NHEADS*S_LEN*HDIM*2);    // colpack
  bf16*  Oc    = (bf16*) carve((size_t)S_LEN*OCAT*2);           // packed-A
  float* Oproj = (float*)carve((size_t)S_LEN*HDIM*4);
  bf16*  yb    = (bf16*) carve((size_t)S_LEN*HDIM*2);           // packed-A
  float* GI    = (float*)carve((size_t)S_LEN*GDIM*4);

  embed_kernel<<<S_LEN, HDIM, 0, stream>>>(tokens, emb, x, xb);

  pack_b_kernel<<<dim3((HDIM*HDIM)/256, 1, NHEADS), 256, 0, stream>>>(
      Wq, Wq_b, HDIM, HDIM, 0, (long)HDIM*HDIM, (long)HDIM*HDIM);
  pack_b_kernel<<<dim3((HDIM*HDIM)/256, 1, NHEADS), 256, 0, stream>>>(
      Wk, Wk_b, HDIM, HDIM, 0, (long)HDIM*HDIM, (long)HDIM*HDIM);
  pack_b_kernel<<<dim3((HDIM*HDIM)/256, 1, NHEADS), 256, 0, stream>>>(
      Wv, Wv_b, HDIM, HDIM, 0, (long)HDIM*HDIM, (long)HDIM*HDIM);
  pack_b_kernel<<<dim3((OCAT*HDIM)/256, 1, 1), 256, 0, stream>>>(
      Wo, Wo_b, OCAT, HDIM, 0, 0L, 0L);
  pack_b_kernel<<<dim3((HDIM*GDIM)/256, 1, 1), 256, 0, stream>>>(
      W_ih, WihT, HDIM, GDIM, 1, 0L, 0L);   // B = W_ih^T

  dim3 gqkv(HDIM/64, S_LEN/64, NHEADS);
  gemm_packed_kernel<<<gqkv, 128, 0, stream>>>(xb, Wq_b, bq, Qb,
      S_LEN, HDIM, HDIM, 0L, (long)HDIM*HDIM, (long)HDIM, (long)S_LEN*HDIM, 1);
  gemm_packed_kernel<<<gqkv, 128, 0, stream>>>(xb, Wk_b, bk, Kb,
      S_LEN, HDIM, HDIM, 0L, (long)HDIM*HDIM, (long)HDIM, (long)S_LEN*HDIM, 1);
  gemm_packed_kernel<<<gqkv, 128, 0, stream>>>(xb, Wv_b, bv, Vb,
      S_LEN, HDIM, HDIM, 0L, (long)HDIM*HDIM, (long)HDIM, (long)S_LEN*HDIM, 2);

  const int attn_smem = (3*16384 + 4096) * (int)sizeof(bf16);   // 104 KB
  (void)hipFuncSetAttribute(reinterpret_cast<const void*>(attn_kernel),
                            hipFuncAttributeMaxDynamicSharedMemorySize, attn_smem);
  dim3 gattn(S_LEN/64, NHEADS);
  attn_kernel<<<gattn, 128, attn_smem, stream>>>(Qb, Kb, Vb, Oc);

  dim3 gop(HDIM/64, S_LEN/64, 1);
  gemm_packed_kernel<<<gop, 128, 0, stream>>>(Oc, Wo_b, bo, Oproj,
      S_LEN, HDIM, OCAT, 0L, 0L, 0L, 0L, 0);

  ln_kernel<<<S_LEN, HDIM, 0, stream>>>(x, Oproj, ln_g, ln_b, yb);

  dim3 ggi(GDIM/64, S_LEN/64, 1);
  gemm_packed_kernel<<<ggi, 128, 0, stream>>>(yb, WihT, b_ih, GI,
      S_LEN, GDIM, HDIM, 0L, 0L, 0L, 0L, 0);

  gru_kernel<<<1, GDIM, 0, stream>>>(GI, W_hh, b_hh, (float*)d_out);
}
